// DA_conv_75849122448156
// MI455X (gfx1250) — matrix operand order repaired and checked by reading, compiled, not run on hardware
//
#include <hip/hip_runtime.h>
#include <hip/hip_bf16.h>

typedef __attribute__((ext_vector_type(16))) _Float16 v16h;
typedef __attribute__((ext_vector_type(8)))  _Float16 v8h;
typedef __attribute__((ext_vector_type(8)))  float    v8f;

#define BATCH 16
#define CH    64
#define IMH   192
#define IMW   192
#define TH    4            // output rows per block
#define XROWS (TH + 2)     // halo rows in LDS
#define WP    (IMW + 2)    // padded width (w=-1 and w=192 zero columns)
#define NTILES (TH * IMW / 16)   // 48 N-tiles (16 pixels each) per block

// ---------------------------------------------------------------------------
// Prep: dynamic-kernel MLP (f32) + conv_w -> f16 repack [tap][co][ci]
// ---------------------------------------------------------------------------
__global__ __launch_bounds__(256) void prep_kernel(
    const float* __restrict__ k_v, const float* __restrict__ W1,
    const float* __restrict__ W2,  const float* __restrict__ conv_w,
    float* __restrict__ kern_out,  _Float16* __restrict__ wh_out)
{
    __shared__ float hid[BATCH * 64];
    const int t = threadIdx.x;

    // hid[b][j] = leaky_relu(sum_i k_v[b][i] * W1[j][i])
    for (int idx = t; idx < BATCH * 64; idx += 256) {
        int b = idx >> 6, j = idx & 63;
        float s = 0.f;
        for (int i = 0; i < 64; ++i) s += k_v[b * 64 + i] * W1[j * 64 + i];
        hid[idx] = (s > 0.f) ? s : 0.1f * s;
    }
    __syncthreads();

    // kern[b][oc] = sum_j hid[b][j] * W2[oc][j],  oc = c*9 + kh*3 + kw
    for (int idx = t; idx < BATCH * CH * 9; idx += 256) {
        int b = idx / (CH * 9), oc = idx % (CH * 9);
        float s = 0.f;
        for (int j = 0; j < 64; ++j) s += hid[b * 64 + j] * W2[oc * 64 + j];
        kern_out[idx] = s;
    }

    // wh[tap][co][ci] = f16(conv_w[co][ci][kh][kw]), tap = kh*3+kw
    for (int idx = t; idx < 9 * CH * CH; idx += 256) {
        int tap = idx / (CH * CH);
        int co  = (idx / CH) % CH;
        int ci  = idx % CH;
        wh_out[idx] = (_Float16)conv_w[(co * CH + ci) * 9 + tap];
    }
}

// ---------------------------------------------------------------------------
// Fused dense conv (WMMA implicit GEMM) + dynamic depthwise + leaky + add
// grid = (BATCH, IMH/TH), block = 256 (8 waves)
// Wave -> (co-tile, ntile-group): each wave holds its 18 A operands (144 VGPRs)
// in registers for the whole block, so the hot loop has zero A-memory ops and
// nothing for LICM to hoist/spill. One v8f accumulator live at a time.
// ---------------------------------------------------------------------------
__global__ __launch_bounds__(256, 1) void fused_conv_kernel(
    const float* __restrict__ x, const _Float16* __restrict__ wh,
    const float* __restrict__ kern, const float* __restrict__ conv_b,
    float* __restrict__ out)
{
    __shared__ __align__(32) _Float16 xs[XROWS][WP][CH]; // ~145.5 KB, f16, [row][col][ci]
    __shared__ float kds[CH * 9];                        // dynamic kernels for this image

    const int n    = blockIdx.x;             // image
    const int h0   = blockIdx.y * TH;        // first output row
    const int t    = threadIdx.x;
    const int wave = t >> 5;
    const int lane = t & 31;
    const int lh   = lane & 15;
    const int ksel = (lane < 16) ? 0 : 1;

    const int mtile = wave >> 1;             // 0..3 : fixed co-tile per wave
    const int grp   = wave & 1;              // 0..1 : ntile interleave group

    // --- A operands for this wave's co-tile: 18 x v16h, kept in registers ---
    // A operand k order used here (same as B): lane holds M=lh, K = {ci0..ci0+15} (ci0 = kc*32 + ksel*16)
    v16h aregs[18];
    #pragma unroll
    for (int tap = 0; tap < 9; ++tap) {
        #pragma unroll
        for (int kc = 0; kc < 2; ++kc) {
            const _Float16* ap =
                wh + tap * CH * CH + (mtile * 16 + lh) * CH + kc * 32 + ksel * 16;
            union { v16h v; v8h h[2]; } ua;
            ua.h[0] = *(const v8h*)ap;
            ua.h[1] = *(const v8h*)(ap + 8);
            aregs[tap * 2 + kc] = ua.v;
        }
    }

    // --- stage dynamic kernels ---
    for (int i = t; i < CH * 9; i += 256) kds[i] = kern[n * CH * 9 + i];

    // --- zero the w-pad columns ---
    for (int i = t; i < XROWS * CH; i += 256) {
        int r = i / CH, c = i % CH;
        xs[r][0][c]      = (_Float16)0.f;
        xs[r][WP - 1][c] = (_Float16)0.f;
    }
    // --- stage x tile (f32 -> f16, NCHW -> [row][col][ci]) with H halo ---
    for (int i = t; i < CH * XROWS * IMW; i += 256) {
        int c   = i / (XROWS * IMW);
        int rem = i % (XROWS * IMW);
        int r   = rem / IMW;
        int w   = rem % IMW;
        int h   = h0 - 1 + r;
        float v = (h >= 0 && h < IMH) ? x[((n * CH + c) * IMH + h) * IMW + w] : 0.f;
        xs[r][w + 1][c] = (_Float16)v;
    }
    __syncthreads();

    // --- explicitly hoist nt-invariant finalize data (prevents LICM spills) ---
    float kreg[8][9];   // dynamic 3x3 taps for this lane's 8 output channels
    float bias[8];
    #pragma unroll
    for (int e = 0; e < 8; ++e) {
        const int co = mtile * 16 + e + ksel * 8;
        bias[e] = conv_b[co];
        #pragma unroll
        for (int q = 0; q < 9; ++q) kreg[e][q] = kds[co * 9 + q];
    }

    // --- WMMA sweep: this wave's co-tile over its 24 N-tiles ---
    #pragma unroll 1
    for (int nt = grp; nt < NTILES; nt += 2) {
        const int r  = nt / (IMW / 16);          // local output row 0..TH-1
        const int w0 = (nt % (IMW / 16)) * 16;   // output col base

        v8f acc = {0.f, 0.f, 0.f, 0.f, 0.f, 0.f, 0.f, 0.f};

        #pragma unroll
        for (int tap = 0; tap < 9; ++tap) {
            const int kh = tap / 3, kw = tap % 3;
            #pragma unroll
            for (int kc = 0; kc < 2; ++kc) {
                // B operand: lane holds N = w0+lh, 16 contiguous K = ci values
                v16h bv = *(const v16h*)&xs[r + kh][w0 + lh + kw][kc * 32 + ksel * 16];
                acc = __builtin_amdgcn_wmma_f32_16x16x32_f16(
                    false, aregs[tap * 2 + kc], false, bv, (short)0, acc, false, false);
            }
        }

        // --- finalize: depthwise (VALU) + leaky + bias + dense acc ---
        const int wcol = w0 + lh;        // output col (N = lane%16)
        const int hrow = h0 + r;
        #pragma unroll
        for (int e = 0; e < 8; ++e) {
            const int co = mtile * 16 + e + ksel * 8;  // C/D layout: M = e (+8 hi lanes)
            float dw = 0.f;
            #pragma unroll
            for (int kh2 = 0; kh2 < 3; ++kh2)
                #pragma unroll
                for (int kw2 = 0; kw2 < 3; ++kw2)
                    dw += kreg[e][kh2 * 3 + kw2] *
                          (float)xs[r + kh2][wcol + kw2][co];
            dw = (dw > 0.f) ? dw : 0.1f * dw;
            out[((n * CH + co) * IMH + hrow) * IMW + wcol] = dw + acc[e] + bias[e];
        }
    }
}

// ---------------------------------------------------------------------------
extern "C" void kernel_launch(void* const* d_in, const int* in_sizes, int n_in,
                              void* d_out, int out_size, void* d_ws, size_t ws_size,
                              hipStream_t stream) {
    const float* x      = (const float*)d_in[0];
    const float* k_v    = (const float*)d_in[1];
    const float* W1     = (const float*)d_in[2];
    const float* W2     = (const float*)d_in[3];
    const float* conv_w = (const float*)d_in[4];
    const float* conv_b = (const float*)d_in[5];
    float* out = (float*)d_out;

    // workspace layout: kern f32 (16*64*9), then wh f16 (9*64*64)
    float*    kern = (float*)d_ws;
    _Float16* wh   = (_Float16*)((char*)d_ws + BATCH * CH * 9 * sizeof(float));

    prep_kernel<<<1, 256, 0, stream>>>(k_v, W1, W2, conv_w, kern, wh);

    dim3 grid(BATCH, IMH / TH);
    fused_conv_kernel<<<grid, 256, 0, stream>>>(x, wh, kern, conv_b, out);
}
